// CentroidDistance_22711787062022
// MI455X (gfx1250) — matrix operand order repaired and checked by reading, not compiled, not run on hardware
//
#include <hip/hip_runtime.h>
#include <hip/hip_bf16.h>
#include <math.h>

// Problem constants (from reference): N=20000 nodes, C=1024 centroids, D=256.
#define NN 20000
#define CC 1024
#define DD 256
#define PEPS 1e-5f

typedef __attribute__((ext_vector_type(16))) __bf16 v16bf;
typedef __attribute__((ext_vector_type(8)))  float  v8f;

union F16Q { float4 q[4]; float f[16]; };

// Split 16 f32 values into bf16 hi/lo fragments (x ~= hi + lo, each RN-rounded).
__device__ __forceinline__ void split16(const float* __restrict__ f, v16bf& hi, v16bf& lo) {
#pragma unroll
  for (int i = 0; i < 16; ++i) {
    float x = f[i];
    __bf16 h = (__bf16)x;
    hi[i] = h;
    lo[i] = (__bf16)(x - (float)h);
  }
}

// Row-wise squared L2 norm of a [rows, 256] row-major matrix.
__global__ void rownorm2_kernel(const float* __restrict__ M, float* __restrict__ out, int rows) {
  int r = blockIdx.x * blockDim.x + threadIdx.x;
  if (r >= rows) return;
  const float4* p = (const float4*)(M + (size_t)r * DD);
  float s = 0.f;
#pragma unroll 8
  for (int i = 0; i < DD / 4; ++i) {
    float4 v = p[i];
    s += v.x * v.x + v.y * v.y + v.z * v.z + v.w * v.w;
  }
  out[r] = s;
}

// Fused Poincare distance:
//   xc  = X @ C^T via split-bf16 WMMA (hi*hi + hi*lo + lo*hi), f32 accumulate
//   d   = acosh(1 + 2*max(x2+c2-2xc,0)/denom) * mask
// Block: 256 threads (8 waves). Block tile 128x64, wave tile 32x32 (2x2 WMMA tiles).
__global__ __launch_bounds__(256)
void poincare_wmma_kernel(const float* __restrict__ X, const float* __restrict__ Cmat,
                          const float* __restrict__ mask, const float* __restrict__ x2a,
                          const float* __restrict__ c2a, float* __restrict__ out)
{
  const int lane = threadIdx.x & 31;
  const int wave = threadIdx.x >> 5;
  const int half = lane >> 4;   // which half-wave (affects K slice / M row group)
  const int lrow = lane & 15;   // row (A) / col (B,C,D) within 16

  const int rowBase = blockIdx.y * 128 + (wave & 3) * 32;   // 4 waves down M
  const int colBase = blockIdx.x * 64  + (wave >> 2) * 32;  // 2 waves across N

  v8f acc[2][2] = {};

  for (int k0 = 0; k0 < DD; k0 += 32) {
    v16bf ah[2], al[2], bh[2], bl[2];

    // ---- A fragments (16x32 bf16): lane l&15 = row M.
    // half 0 holds K k0+0..15; half 1 holds K k0+16..31.
    // OOB rows clamped (keeps EXEC all-ones for WMMA); stores predicated later.
#pragma unroll
    for (int mt = 0; mt < 2; ++mt) {
      int m = rowBase + mt * 16 + lrow;
      m = (m < NN) ? m : (NN - 1);
      const float* p = X + (size_t)m * DD + k0 + half * 16;
      F16Q t;
      t.q[0] = ((const float4*)p)[0];
      t.q[1] = ((const float4*)p)[1];
      t.q[2] = ((const float4*)p)[2];
      t.q[3] = ((const float4*)p)[3];
      split16(t.f, ah[mt], al[mt]);
    }

    // ---- B fragments (32x16 bf16): B[k][n] = centroids[n][k] (row-major cent).
    // lane lrow = column n; half 0 holds K k0+0..15, half 1 holds K k0+16..31
    // -> 16 contiguous f32 per lane.
#pragma unroll
    for (int nt = 0; nt < 2; ++nt) {
      int c = colBase + nt * 16 + lrow;             // always < 1024
      const float* p = Cmat + (size_t)c * DD + k0 + half * 16;
      F16Q t;
      t.q[0] = ((const float4*)p)[0];
      t.q[1] = ((const float4*)p)[1];
      t.q[2] = ((const float4*)p)[2];
      t.q[3] = ((const float4*)p)[3];
      split16(t.f, bh[nt], bl[nt]);
    }

    // ---- 12 WMMAs per K-step: hi*hi + hi*lo + lo*hi into the same f32 acc.
#pragma unroll
    for (int mt = 0; mt < 2; ++mt)
#pragma unroll
      for (int nt = 0; nt < 2; ++nt) {
        acc[mt][nt] = __builtin_amdgcn_wmma_f32_16x16x32_bf16(
            false, ah[mt], false, bh[nt], (short)0, acc[mt][nt], false, false);
        acc[mt][nt] = __builtin_amdgcn_wmma_f32_16x16x32_bf16(
            false, ah[mt], false, bl[nt], (short)0, acc[mt][nt], false, false);
        acc[mt][nt] = __builtin_amdgcn_wmma_f32_16x16x32_bf16(
            false, al[mt], false, bh[nt], (short)0, acc[mt][nt], false, false);
      }
  }

  // ---- Fused Poincare epilogue. C/D layout: elem r, lanes0-15 -> M=r,
  // lanes16-31 -> M=r+8; N = lane&15.
#pragma unroll
  for (int mt = 0; mt < 2; ++mt) {
    const int rbase = rowBase + mt * 16 + half * 8;
    float x2v[8], mkv[8];
#pragma unroll
    for (int r = 0; r < 8; ++r) {
      int row = rbase + r;
      row = (row < NN) ? row : (NN - 1);
      x2v[r] = x2a[row];
      mkv[r] = mask[row];
    }
#pragma unroll
    for (int nt = 0; nt < 2; ++nt) {
      const int col = colBase + nt * 16 + lrow;
      const float c2v = c2a[col];
      const float dc  = 1.0f - fminf(c2v, 1.0f - PEPS);
#pragma unroll
      for (int r = 0; r < 8; ++r) {
        int row = rbase + r;
        if (row < NN) {
          float xc    = acc[mt][nt][r];
          float sqd   = fmaxf(x2v[r] + c2v - 2.0f * xc, 0.0f);
          float denom = (1.0f - fminf(x2v[r], 1.0f - PEPS)) * dc;
          // acosh(1+t) = log(1 + t + sqrt(t*(t+2))), t >= 1e-7 (== max(arg,1+1e-7))
          float t     = fmaxf(2.0f * sqd / denom, 1e-7f);
          float dist  = logf(1.0f + t + sqrtf(t * (t + 2.0f)));
          out[(size_t)row * CC + col] = dist * mkv[r];
        }
      }
    }
  }
}

extern "C" void kernel_launch(void* const* d_in, const int* in_sizes, int n_in,
                              void* d_out, int out_size, void* d_ws, size_t ws_size,
                              hipStream_t stream) {
  const float* node = (const float*)d_in[0];   // [20000, 256]
  const float* mask = (const float*)d_in[1];   // [20000, 1]
  const float* cent = (const float*)d_in[2];   // [1024, 256]
  float* out = (float*)d_out;                  // [20000, 1024]

  // Workspace: x2[N] then c2[C] (256B aligned). Total < 96 KB.
  float* x2 = (float*)d_ws;
  float* c2 = (float*)((char*)d_ws + (((size_t)NN * sizeof(float) + 255) & ~(size_t)255));

  rownorm2_kernel<<<(NN + 255) / 256, 256, 0, stream>>>(node, x2, NN);
  rownorm2_kernel<<<(CC + 255) / 256, 256, 0, stream>>>(cent, c2, CC);

  dim3 grid(CC / 64, (NN + 127) / 128);  // (16, 157)
  poincare_wmma_kernel<<<grid, 256, 0, stream>>>(node, cent, mask, x2, c2, out);
}
